// FourLayer_64F_56246891708648
// MI455X (gfx1250) — compile-verified, matrix operand order checked by reading, not run on hardware
//
#include <hip/hip_runtime.h>
#include <hip/hip_bf16.h>

// ---------------------------------------------------------------------------
// CDNA5 (gfx1250) implementation of the 4-layer conv backbone + Wasserstein
// episode similarity.  Conv layers 2-4 and the covariance Gram matrices run
// as implicit GEMMs on v_wmma_f32_16x16x32_f16 (wave32 WMMA, f32 accumulate,
// BN folded into per-channel scale/bias epilogue).  Bulk global->LDS staging
// uses the Tensor Data Mover (tensor_load_to_lds + s_wait_tensorcnt) when the
// builtin is available, with a cooperative-copy fallback.
// ---------------------------------------------------------------------------

typedef __attribute__((ext_vector_type(16))) _Float16 v16h;
typedef __attribute__((ext_vector_type(8)))  _Float16 v8h;
typedef __attribute__((ext_vector_type(8)))  float    v8f;
typedef __attribute__((ext_vector_type(4)))  unsigned int v4u;
typedef __attribute__((ext_vector_type(8)))  int v8i;
typedef __attribute__((ext_vector_type(4)))  int v4i;

__device__ __forceinline__ v16h cat16(v8h lo, v8h hi) {
  return __builtin_shufflevector(lo, hi, 0, 1, 2, 3, 4, 5, 6, 7,
                                 8, 9, 10, 11, 12, 13, 14, 15);
}

// ------------------------------ constants ----------------------------------
#define NQ        300
#define NS        20
#define NIMG      320
#define NEG_SLOPE 0.2f
#define EPS_BN    1e-5f
// packed weight block: [tap 9][kc 2][ntile 4][lane 32][j 16] f16
#define WPACK_HALVES (9 * 2 * 4 * 32 * 16)     // 36864 halves
#define WPACK_BYTES  (WPACK_HALVES * 2)        // 73728 bytes (dynamic LDS)
// swizzled act4 block: [tile 4][kstep 14][lane 32][j 16] f16 per image
#define SWZ_HALVES   (4 * 14 * 32 * 16)        // 28672 halves
#define SWZ_BYTES    (SWZ_HALVES * 2)          // 57344 bytes (dynamic LDS)

// row mapping shared by the A- and B-fragment lane layouts (16-bit WMMA):
// lane<16 holds K {0..7,16..23}, lane>=16 holds K {8..15,24..31}.
__device__ __forceinline__ int frag_row(int j, int hi) {
  return (j < 8) ? (j + hi * 8) : (j + 8 + hi * 8);
}

// ---------------------------------------------------------------------------
// Tensor Data Mover: DMA a flat block of nElem8 * 8 bytes from global memory
// to LDS offset ldsAddr.  D# built per CDNA5 ISA ch.8 (group0/group1; 2D tile
// of one row).  Kernels using this have no static LDS, so the dynamic-LDS
// block starts at offset 0.
// ---------------------------------------------------------------------------
#if __has_builtin(__builtin_amdgcn_tensor_load_to_lds)
#define HAVE_TDM 1
__device__ __forceinline__ void tdm_copy_to_lds(const void* gptr,
                                                unsigned ldsAddr,
                                                unsigned nElem8) {
  unsigned long long ga = (unsigned long long)gptr;
  v4u g0 = {};
  g0[0] = 1u;                                   // count=1, no gather
  g0[1] = ldsAddr;                              // lds_addr
  g0[2] = (unsigned)ga;                         // global_addr[31:0]
  g0[3] = (unsigned)((ga >> 32) & 0x1FFFFFFu)   // global_addr[56:32]
          | (2u << 30);                         // type = 2 ("image")
  v8i g1 = {};
  g1[0] = (int)(3u << 16);                      // data_size = 8 bytes
  g1[1] = (int)((nElem8 & 0xFFFFu) << 16);      // tensor_dim0 lo
  g1[2] = (int)(((nElem8 >> 16) & 0xFFFFu)      // tensor_dim0 hi
                | (1u << 16));                  // tensor_dim1 = 1
  g1[3] = (int)((nElem8 & 0xFFFFu) << 16);      // tile_dim0
  g1[4] = 1;                                    // tile_dim1 = 1
  g1[5] = (int)nElem8;                          // tensor_dim0_stride lo
  v4i g2 = {}, g3 = {};
#if defined(__clang_major__) && (__clang_major__ >= 23)
  v8i gx = {};
  __builtin_amdgcn_tensor_load_to_lds(g0, g1, g2, g3, gx, 0);
#else
  __builtin_amdgcn_tensor_load_to_lds(g0, g1, g2, g3, 0);
#endif
}
#else
#define HAVE_TDM 0
#endif

// ---------------------------------------------------------------------------
// BN fold: scale = g / sqrt(v + eps), bias = b - m * scale   (4 layers x 64)
// ---------------------------------------------------------------------------
__global__ void bnprep_kernel(const float* __restrict__ g,
                              const float* __restrict__ b,
                              const float* __restrict__ m,
                              const float* __restrict__ v,
                              float* __restrict__ scale,
                              float* __restrict__ bias) {
  int t = threadIdx.x;  // 0..255 = 4 layers * 64 ch
  float inv = g[t] * rsqrtf(v[t] + EPS_BN);
  scale[t] = inv;
  bias[t]  = b[t] - m[t] * inv;
}

// ---------------------------------------------------------------------------
// Pack OIHW f32 weights (64,64,3,3) into WMMA-B-fragment-ready f16 layout.
// Fragment element j of lane n holds B[K = frag_row(j, n>=16)][N = n & 15].
// ---------------------------------------------------------------------------
__global__ void pack_kernel(const float* __restrict__ w,
                            _Float16* __restrict__ dst) {
  int tid = blockIdx.x * 256 + threadIdx.x;
  if (tid >= 9 * 2 * 4 * 32) return;
  int lane = tid & 31;
  int rest = tid >> 5;
  int nt   = rest & 3;
  int kc   = (rest >> 2) & 1;
  int tap  = rest >> 3;
  int ky = tap / 3, kx = tap % 3;
  int oc = nt * 16 + (lane & 15);
  int hi = lane >> 4;
#pragma unroll
  for (int j = 0; j < 16; ++j) {
    int ic = kc * 32 + frag_row(j, hi);
    float val = w[((size_t)(oc * 64 + ic) * 3 + ky) * 3 + kx];
    dst[(((size_t)(tap * 2 + kc) * 4 + nt) * 32 + lane) * 16 + j] =
        (_Float16)val;
  }
}

// ---------------------------------------------------------------------------
// Zero a buffer in 16-byte chunks (grid-stride).
// ---------------------------------------------------------------------------
__global__ void zero16_kernel(_Float16* __restrict__ p, unsigned long long n8) {
  v8h z = {};
  v8h* d = (v8h*)p;
  for (unsigned long long i = (unsigned long long)blockIdx.x * 256 + threadIdx.x;
       i < n8; i += (unsigned long long)gridDim.x * 256)
    d[i] = z;
}

// ---------------------------------------------------------------------------
// Layer 1: direct conv 3->64 (3x3 SAME) + BN + LeakyReLU + 2x2 maxpool.
// Writes f16 into the interior of padded act1 [img][44][44][64].
// grid (42*42, nImg), block 64 (out channel).
// ---------------------------------------------------------------------------
__global__ void conv1_kernel(const float* __restrict__ x,
                             const float* __restrict__ w1,
                             const float* __restrict__ scale,
                             const float* __restrict__ bias,
                             _Float16* __restrict__ act1p, int imgBase) {
  int sp  = blockIdx.x;
  int img = blockIdx.y;
  int oc  = threadIdx.x;
  int oy = sp / 42, ox = sp % 42;
  const float* xi = x + (size_t)img * 3 * 84 * 84;
  const float s = scale[oc], bb = bias[oc];
  float best = -1e30f;
#pragma unroll
  for (int cy = 0; cy < 2; ++cy)
#pragma unroll
    for (int cx = 0; cx < 2; ++cx) {
      int Y = oy * 2 + cy, X = ox * 2 + cx;
      float acc = 0.f;
      for (int ic = 0; ic < 3; ++ic)
#pragma unroll
        for (int ky = 0; ky < 3; ++ky) {
          int iy = Y + ky - 1;
          if (iy < 0 || iy >= 84) continue;
#pragma unroll
          for (int kx = 0; kx < 3; ++kx) {
            int ix = X + kx - 1;
            if (ix < 0 || ix >= 84) continue;
            acc += xi[((size_t)ic * 84 + iy) * 84 + ix] *
                   w1[((oc * 3 + ic) * 3 + ky) * 3 + kx];
          }
        }
      float v = acc * s + bb;
      v = (v >= 0.f) ? v : NEG_SLOPE * v;
      best = fmaxf(best, v);
    }
  size_t gi = (size_t)imgBase + img;
  act1p[((gi * 44 + (oy + 1)) * 44 + (ox + 1)) * 64 + oc] = (_Float16)best;
}

// ---------------------------------------------------------------------------
// Implicit-GEMM 3x3 SAME conv, 64->64 channels, on WMMA f32_16x16x32_f16.
//  - input : padded f16 activations [img][H+2][W+2][64] (zero borders)
//  - B     : fragment-ready packed weights, staged in LDS via TDM (72 KB)
//  - epilogue: BN scale/bias + LeakyReLU, f16 store (padded or flat layout)
// grid (ceil(H*W/128), nImg), block 256 (8 wave32), dyn LDS = WPACK_BYTES.
// Each wave: one 16(M)x64(N) tile, K = 9 taps * 64 ch = 18 WMMA steps.
// A-fragment rows with m >= M are loaded unguarded (clamped address) and
// their results simply never stored -> branch-free K-loop.
// ---------------------------------------------------------------------------
__global__ __launch_bounds__(256)
void conv_gemm_kernel(const _Float16* __restrict__ actp,
                      const _Float16* __restrict__ wpack,
                      const float* __restrict__ scale,
                      const float* __restrict__ bias,
                      _Float16* __restrict__ out,
                      int H, int W, int outPadded) {
  extern __shared__ _Float16 wlds[];
  const int tid  = threadIdx.x;
  const int wave = tid >> 5;
  const int lane = tid & 31;

#if HAVE_TDM
  if (wave == 0) {
    tdm_copy_to_lds(wpack, 0u, WPACK_BYTES / 8);
    __builtin_amdgcn_s_wait_tensorcnt(0);
  }
#else
  {
    const v8h* src = (const v8h*)wpack;
    v8h* dst = (v8h*)wlds;
    for (int i = tid; i < WPACK_HALVES / 8; i += 256) dst[i] = src[i];
  }
#endif
  __syncthreads();

  const int img = blockIdx.y;
  const int M   = H * W;
  const int m0  = (blockIdx.x * 8 + wave) * 16;
  if (m0 >= M) return;

  const int r  = lane & 15;
  const int hi = lane >> 4;
  const int m  = m0 + r;
  const int mc = (m < M) ? m : (M - 1);      // clamp: rows >= M are discarded
  const int y  = mc / W;
  const int x  = mc % W;
  const int Wp = W + 2;
  const _Float16* abase =
      actp + (((size_t)img * (H + 2) + (y + 1)) * Wp + (x + 1)) * 64;

  v8f acc[4] = {};

#pragma unroll
  for (int tap = 0; tap < 9; ++tap) {
    const int dy = tap / 3 - 1;
    const int dx = tap % 3 - 1;
    const _Float16* ap = abase + (dy * Wp + dx) * 64;
    if (tap < 8) {
      const int nt2 = tap + 1;
      __builtin_prefetch(abase + ((nt2 / 3 - 1) * Wp + (nt2 % 3 - 1)) * 64,
                         0, 0);
    }
#pragma unroll
    for (int kc = 0; kc < 2; ++kc) {
      const _Float16* p = ap + kc * 32 + hi * 8;
      v8h lo = *(const v8h*)(p);        // K 0..7   (or 8..15)
      v8h hh = *(const v8h*)(p + 16);   // K 16..23 (or 24..31)
      v16h afrag = cat16(lo, hh);
      const _Float16* wb = wlds + (size_t)((tap * 2 + kc) * 4) * 32 * 16;
#pragma unroll
      for (int nt = 0; nt < 4; ++nt) {
        const v8h* q = (const v8h*)(wb + ((size_t)nt * 32 + lane) * 16);
        v16h bfrag = cat16(q[0], q[1]);
        acc[nt] = __builtin_amdgcn_wmma_f32_16x16x32_f16(
            false, afrag, false, bfrag, (short)0, acc[nt], false, false);
      }
    }
  }

#pragma unroll
  for (int nt = 0; nt < 4; ++nt) {
    const int c = nt * 16 + r;
    const float s = scale[c];
    const float bb = bias[c];
#pragma unroll
    for (int j = 0; j < 8; ++j) {
      const int mj = m0 + j + hi * 8;
      if (mj >= M) continue;
      float v = acc[nt][j] * s + bb;
      v = (v >= 0.f) ? v : NEG_SLOPE * v;
      const _Float16 hv = (_Float16)v;
      if (outPadded) {
        const int yy = mj / W, xx = mj % W;
        out[(((size_t)img * (H + 2) + (yy + 1)) * Wp + (xx + 1)) * 64 + c] = hv;
      } else {
        out[((size_t)img * M + mj) * 64 + c] = hv;
      }
    }
  }
}

// ---------------------------------------------------------------------------
// 2x2 maxpool: t2 [img][42*42][64] -> padded act2 [img][23][23][64] interior.
// ---------------------------------------------------------------------------
__global__ void pool_kernel(const _Float16* __restrict__ t2,
                            _Float16* __restrict__ act2p) {
  int sp = blockIdx.x;   // 0..440
  int img = blockIdx.y;
  int c = threadIdx.x;
  int y = sp / 21, x = sp % 21;
  const _Float16* base = t2 + (size_t)img * 1764 * 64;
  float best = -1e30f;
#pragma unroll
  for (int cy = 0; cy < 2; ++cy)
#pragma unroll
    for (int cx = 0; cx < 2; ++cx) {
      float v = (float)base[(((size_t)(2 * y + cy)) * 42 + (2 * x + cx)) * 64 + c];
      best = fmaxf(best, v);
    }
  act2p[(((size_t)img * 23 + (y + 1)) * 23 + (x + 1)) * 64 + c] = (_Float16)best;
}

// ---------------------------------------------------------------------------
// Swizzle act4 [img][441][64] into fragment-ready layout (rows >= 441 are
// zero-padded, K padded to 448).  One buffer serves BOTH WMMA operands since
// the A/B lane layouts are mirrors with the same row function.
// grid (NIMG), block 256.
// ---------------------------------------------------------------------------
__global__ void swz_kernel(const _Float16* __restrict__ act4,
                           _Float16* __restrict__ swz) {
  const int img = blockIdx.x;
  const _Float16* A = act4 + (size_t)img * 441 * 64;
  _Float16* D = swz + (size_t)img * SWZ_HALVES;
  for (int idx = threadIdx.x; idx < 4 * 14 * 32; idx += 256) {
    int lane = idx & 31;
    int ks   = (idx >> 5) % 14;
    int t    = idx / (14 * 32);
    int r = lane & 15, hi = lane >> 4;
    int c = t * 16 + r;
    v16h tv = {};
#pragma unroll
    for (int j = 0; j < 16; ++j) {
      int row = ks * 32 + frag_row(j, hi);
      if (row < 441) tv[j] = A[(size_t)row * 64 + c];
    }
    v8h lo = __builtin_shufflevector(tv, tv, 0, 1, 2, 3, 4, 5, 6, 7);
    v8h hh = __builtin_shufflevector(tv, tv, 8, 9, 10, 11, 12, 13, 14, 15);
    v8h* dst = (v8h*)(D + (size_t)idx * 16);
    dst[0] = lo;
    dst[1] = hh;
  }
}

// ---------------------------------------------------------------------------
// Per-image uncentered Gram G = A^T A (64x64 f32) from the swizzled act4,
// via WMMA f32_16x16x32_f16 over K=448 (zero padded).  Whole per-image block
// (56 KB) staged in LDS via TDM; inner loop is ds_load_b128 + v_wmma only.
// One workgroup (8 waves) per image; 16 tiles -> 2 tiles per wave.
// grid (NIMG), block 256, dyn LDS = SWZ_BYTES.
// ---------------------------------------------------------------------------
__global__ __launch_bounds__(256)
void gram_kernel(const _Float16* __restrict__ swz, float* __restrict__ gram) {
  extern __shared__ _Float16 slds[];
  const int img  = blockIdx.x;
  const int tid  = threadIdx.x;
  const int wave = tid >> 5;
  const int lane = tid & 31;

#if HAVE_TDM
  if (wave == 0) {
    tdm_copy_to_lds(swz + (size_t)img * SWZ_HALVES, 0u, SWZ_BYTES / 8);
    __builtin_amdgcn_s_wait_tensorcnt(0);
  }
#else
  {
    const v8h* src = (const v8h*)(swz + (size_t)img * SWZ_HALVES);
    v8h* dst = (v8h*)slds;
    for (int i = tid; i < SWZ_HALVES / 8; i += 256) dst[i] = src[i];
  }
#endif
  __syncthreads();

  const int r  = lane & 15;
  const int hi = lane >> 4;

  for (int t = wave; t < 16; t += 8) {
    const int mt = t >> 2, nt = t & 3;
    v8f acc = {};
    for (int ks = 0; ks < 14; ++ks) {
      const v8h* qa = (const v8h*)(slds + ((size_t)(mt * 14 + ks) * 32 + lane) * 16);
      const v8h* qb = (const v8h*)(slds + ((size_t)(nt * 14 + ks) * 32 + lane) * 16);
      v16h af = cat16(qa[0], qa[1]);
      v16h bf = cat16(qb[0], qb[1]);
      acc = __builtin_amdgcn_wmma_f32_16x16x32_f16(
          false, af, false, bf, (short)0, acc, false, false);
    }
#pragma unroll
    for (int j = 0; j < 8; ++j) {
      int rowm = mt * 16 + j + hi * 8;
      gram[(size_t)img * 4096 + (size_t)rowm * 64 + nt * 16 + r] = acc[j];
    }
  }
}

// ---------------------------------------------------------------------------
// Per-image per-channel sums (for means).  grid NIMG, block 64.
// ---------------------------------------------------------------------------
__global__ void sums_kernel(const _Float16* __restrict__ act4,
                            float* __restrict__ sums) {
  int img = blockIdx.x;
  int c = threadIdx.x;
  const _Float16* A = act4 + (size_t)img * 441 * 64 + c;
  float s = 0.f;
  for (int n = 0; n < 441; ++n) s += (float)A[(size_t)n * 64];
  sums[img * 64 + c] = s;
}

// ---------------------------------------------------------------------------
// Episode similarity.  One block per (episode e, query i, support j).
//  cov = (G - n mu mu^T)/(n-1) reconstructed on the fly; leave-one-out
//  pooled stats come from summed Grams/sums of the other 4 supports.
// ---------------------------------------------------------------------------
__global__ __launch_bounds__(256)
void sim_kernel(const float* __restrict__ gram,
                const float* __restrict__ sums,
                float* __restrict__ out) {
  const int b = blockIdx.x;            // e*375 + i*5 + j
  const int e = b / 375;
  const int rem = b % 375;
  const int i = rem / 5;
  const int j = rem % 5;
  const int qimg = e * 75 + i;
  const int sbase = NQ + e * 5;

  __shared__ float muq[64], mus[64], mul[64];
  __shared__ float red1[256], red2[256], red3[256], red4[256];

  const int t = threadIdx.x;
  if (t < 64) {
    float sq = sums[qimg * 64 + t];
    float ss = sums[(sbase + j) * 64 + t];
    float sl = 0.f;
    for (int k = 0; k < 5; ++k)
      if (k != j) sl += sums[(sbase + k) * 64 + t];
    muq[t] = sq * (1.f / 441.f);
    mus[t] = ss * (1.f / 441.f);
    mul[t] = sl * (1.f / 1764.f);
  }
  __syncthreads();

  float dm1 = 0.f, dm2 = 0.f;
  if (t < 64) {
    float d1 = muq[t] - mus[t];
    float d2 = muq[t] - mul[t];
    dm1 = d1 * d1;
    dm2 = d2 * d2;
  }

  const float* Gq = gram + (size_t)qimg * 4096;
  const float* Gs = gram + (size_t)(sbase + j) * 4096;
  float dc1 = 0.f, dc2 = 0.f;
  for (int idx = t; idx < 4096; idx += 256) {
    int rr = idx >> 6, cc = idx & 63;
    float covq = (Gq[idx] - 441.f * muq[rr] * muq[cc]) * (1.f / 440.f);
    float covs = (Gs[idx] - 441.f * mus[rr] * mus[cc]) * (1.f / 440.f);
    float gl = 0.f;
    for (int k = 0; k < 5; ++k)
      if (k != j) gl += gram[(size_t)(sbase + k) * 4096 + idx];
    float covl = (gl - 1764.f * mul[rr] * mul[cc]) * (1.f / 1763.f);
    float a = covq - covs;
    float bb = covq - covl;
    dc1 += a * a;
    dc2 += bb * bb;
  }

  red1[t] = dm1; red2[t] = dm2; red3[t] = dc1; red4[t] = dc2;
  __syncthreads();
  for (int off = 128; off > 0; off >>= 1) {
    if (t < off) {
      red1[t] += red1[t + off];
      red2[t] += red2[t + off];
      red3[t] += red3[t + off];
      red4[t] += red4[t + off];
    }
    __syncthreads();
  }
  if (t == 0) {
    float dis1 = sqrtf(red1[0]) * (1.f / 64.f) + sqrtf(red3[0]) * (1.f / 4096.f);
    float dis2 = sqrtf(red2[0]) * (1.f / 64.f) + sqrtf(red4[0]) * (1.f / 4096.f);
    out[b] = -dis1 + dis2;
  }
}

// ---------------------------------------------------------------------------
// Host launcher
// ---------------------------------------------------------------------------
extern "C" void kernel_launch(void* const* d_in, const int* in_sizes, int n_in,
                              void* d_out, int out_size, void* d_ws,
                              size_t ws_size, hipStream_t stream) {
  const float* x1 = (const float*)d_in[0];
  const float* x2 = (const float*)d_in[1];
  const float* w1 = (const float*)d_in[2];
  const float* w2 = (const float*)d_in[3];
  const float* w3 = (const float*)d_in[4];
  const float* w4 = (const float*)d_in[5];
  const float* bg = (const float*)d_in[6];
  const float* bb = (const float*)d_in[7];
  const float* bm = (const float*)d_in[8];
  const float* bv = (const float*)d_in[9];
  float* out = (float*)d_out;

  // workspace carve-up (256B aligned)
  char* ws = (char*)d_ws;
  size_t off = 0;
  auto alloc = [&](size_t bytes) -> void* {
    void* p = ws + off;
    off += (bytes + 255) & ~(size_t)255;
    return p;
  };
  float*    scale = (float*)alloc(4 * 64 * sizeof(float));
  float*    bias  = (float*)alloc(4 * 64 * sizeof(float));
  _Float16* wp2   = (_Float16*)alloc(WPACK_BYTES);
  _Float16* wp3   = (_Float16*)alloc(WPACK_BYTES);
  _Float16* wp4   = (_Float16*)alloc(WPACK_BYTES);
  const size_t act1pB = (size_t)NIMG * 44 * 44 * 64 * 2;   // 79.3 MB
  const size_t t2B    = (size_t)NIMG * 1764 * 64 * 2;      // 72.3 MB
  const size_t act2pB = (size_t)NIMG * 23 * 23 * 64 * 2;   // 21.7 MB
  const size_t act4B  = (size_t)NIMG * 441 * 64 * 2;       // 18.1 MB
  const size_t swzB   = (size_t)NIMG * SWZ_BYTES;          // 18.4 MB
  _Float16* act1p = (_Float16*)alloc(act1pB);
  _Float16* t2    = (_Float16*)alloc(t2B);
  _Float16* act2p = (_Float16*)alloc(act2pB);
  _Float16* act3p = (_Float16*)alloc(act2pB);
  _Float16* act4  = (_Float16*)alloc(act4B);
  _Float16* swzb  = (_Float16*)alloc(swzB);
  float*    gramb = (float*)alloc((size_t)NIMG * 4096 * sizeof(float));
  float*    sumsb = (float*)alloc((size_t)NIMG * 64 * sizeof(float));
  (void)ws_size; (void)in_sizes; (void)n_in; (void)out_size;

  // 0) BN fold + weight packing
  bnprep_kernel<<<1, 256, 0, stream>>>(bg, bb, bm, bv, scale, bias);
  pack_kernel<<<9, 256, 0, stream>>>(w2, wp2);
  pack_kernel<<<9, 256, 0, stream>>>(w3, wp3);
  pack_kernel<<<9, 256, 0, stream>>>(w4, wp4);

  // 1) zero padded activation buffers (borders must be 0 for implicit GEMM)
  zero16_kernel<<<4096, 256, 0, stream>>>(act1p, act1pB / 16);
  zero16_kernel<<<4096, 256, 0, stream>>>(act2p, act2pB / 16);
  zero16_kernel<<<4096, 256, 0, stream>>>(act3p, act2pB / 16);

  // 2) layer 1 (direct conv + bn + lrelu + pool) for queries and supports
  conv1_kernel<<<dim3(1764, NQ), 64, 0, stream>>>(x1, w1, scale, bias, act1p, 0);
  conv1_kernel<<<dim3(1764, NS), 64, 0, stream>>>(x2, w1, scale, bias, act1p, NQ);

  // 3) layer 2 GEMM (42x42), then pool -> 21x21
  conv_gemm_kernel<<<dim3(14, NIMG), 256, WPACK_BYTES, stream>>>(
      act1p, wp2, scale + 64, bias + 64, t2, 42, 42, /*outPadded=*/0);
  pool_kernel<<<dim3(441, NIMG), 64, 0, stream>>>(t2, act2p);

  // 4) layers 3 & 4 GEMM (21x21)
  conv_gemm_kernel<<<dim3(4, NIMG), 256, WPACK_BYTES, stream>>>(
      act2p, wp3, scale + 128, bias + 128, act3p, 21, 21, /*outPadded=*/1);
  conv_gemm_kernel<<<dim3(4, NIMG), 256, WPACK_BYTES, stream>>>(
      act3p, wp4, scale + 192, bias + 192, act4, 21, 21, /*outPadded=*/0);

  // 5) per-image second moments via WMMA, then episode similarity
  swz_kernel<<<NIMG, 256, 0, stream>>>(act4, swzb);
  gram_kernel<<<NIMG, 256, SWZ_BYTES, stream>>>(swzb, gramb);
  sums_kernel<<<NIMG, 64, 0, stream>>>(act4, sumsb);
  sim_kernel<<<4 * 75 * 5, 256, 0, stream>>>(gramb, sumsb, out);
}
